// LinearChainCrf_51376398795476
// MI455X (gfx1250) — compile-verified, matrix operand order checked by reading, not run on hardware
//
#include <hip/hip_runtime.h>
#include <hip/hip_bf16.h>

// Linear-chain CRF NLL for MI455X (gfx1250, wave32, WMMA).
// logsumexp recurrence turned into f16 WMMA matmuls against exp(transitions),
// with running per-row shifts for numerical stability.

typedef __attribute__((ext_vector_type(16))) _Float16 v16h;
typedef __attribute__((ext_vector_type(8)))  _Float16 v8h;
typedef __attribute__((ext_vector_type(8)))  float    v8f;

union V16 { v16h v; v8h h[2]; };

#define Bsz  256
#define Tlen 1024
#define Kdim 128

#define LOG2E 1.4426950408889634f
#define LN2   0.6931471805599453f

// raw hardware transcendentals: v_exp_f32 = exp2, v_log_f32 = log2
#if __has_builtin(__builtin_amdgcn_exp2f)
#define FAST_EXP2(x) __builtin_amdgcn_exp2f(x)
#else
#define FAST_EXP2(x) exp2f(x)
#endif
#if __has_builtin(__builtin_amdgcn_logf)
#define FAST_LOG2(x) __builtin_amdgcn_logf(x)
#else
#define FAST_LOG2(x) log2f(x)
#endif

// ---------------------------------------------------------------------------
// Kernel 1: expT16[i*K + j] = (f16) exp(transitions[i][j])
// ---------------------------------------------------------------------------
__global__ void crf_exp_transitions(const float* __restrict__ trans,
                                    _Float16* __restrict__ expT16) {
  int idx = blockIdx.x * blockDim.x + threadIdx.x;
  if (idx < Kdim * Kdim) expT16[idx] = (_Float16)FAST_EXP2(trans[idx] * LOG2E);
}

// ---------------------------------------------------------------------------
// Kernel 2: gold path score, one wave (32 lanes) per batch row
// ---------------------------------------------------------------------------
__global__ void crf_gold(const float* __restrict__ em,
                         const float* __restrict__ trans,
                         const float* __restrict__ startT,
                         const float* __restrict__ endT,
                         const int* __restrict__ tags,
                         const unsigned char* __restrict__ mask,
                         float* __restrict__ gold) {
  int wave = threadIdx.x >> 5;
  int lane = threadIdx.x & 31;
  int b = blockIdx.x * (blockDim.x >> 5) + wave;
  if (b >= Bsz) return;
  const int* tg = tags + (size_t)b * Tlen;
  const unsigned char* mk = mask + (size_t)b * Tlen;
  const float* eb = em + (size_t)b * Tlen * Kdim;

  float part = 0.f;
  int len = 0;
  for (int t = lane; t < Tlen; t += 32) {
    int m = mk[t];
    len += m;
    if (t >= 1 && m) {
      int tp = tg[t - 1], tc = tg[t];
      part += trans[tp * Kdim + tc] + eb[(size_t)t * Kdim + tc];
    }
  }
  for (int d = 16; d >= 1; d >>= 1) {
    part += __shfl_xor(part, d, 32);
    len  += __shfl_xor(len, d, 32);
  }
  if (lane == 0) {
    int first = tg[0];
    float sc = startT[first] + eb[first] + part;
    int last = tg[len - 1];
    gold[b] = sc + endT[last];
  }
}

// ---------------------------------------------------------------------------
// Kernel 3: forward scan with WMMA. 16 WGs x 16 batch rows, 8 waves/WG.
// Wave w owns destination columns [16w, 16w+16). B (expT) is VGPR-resident.
// ---------------------------------------------------------------------------
__launch_bounds__(256, 1)
__global__ void crf_logz(const float* __restrict__ em,
                         const _Float16* __restrict__ expT16,
                         const float* __restrict__ startT,
                         const float* __restrict__ endT,
                         const unsigned char* __restrict__ mask,
                         const float* __restrict__ gold,
                         float* __restrict__ out) {
  // padded rows: 136 f16 = 272B = 68 dwords -> conflict-free strided row reads
  __shared__ __align__(16) _Float16 ldsA[16][136];
  __shared__ __align__(16) float    ldsRed[16][8];
  __shared__ __align__(16) float    ldsF[16][132];

  const int wave = threadIdx.x >> 5;
  const int lane = threadIdx.x & 31;
  const int half = lane >> 4;
  const int n    = lane & 15;
  const int off  = half ? 8 : 0;      // K half-offset (A/B) and M half-offset (C)
  const int b0   = blockIdx.x * 16;
  const int col  = wave * 16 + n;     // destination tag j owned by this lane

  // ---- load B fragments: expT[k][col], k in ISA wave32 16-bit A/B layout ----
  V16 bm[4];
#pragma unroll
  for (int c = 0; c < 4; ++c) {
    _Float16 tmp[16];
#pragma unroll
    for (int e = 0; e < 16; ++e) {
      int k = 32 * c + ((e < 8) ? e : e + 8) + off;
      tmp[e] = expT16[k * Kdim + col];
    }
#pragma unroll
    for (int e = 0; e < 8; ++e) { bm[c].h[0][e] = tmp[e]; bm[c].h[1][e] = tmp[8 + e]; }
  }

  // ---- init alpha (stored in C layout: vgpr r -> batch row r+off, lane n -> col) ----
  const float* eb = em + (size_t)(b0 + off) * Tlen * Kdim + col;
  const unsigned char* mb = mask + (size_t)(b0 + off) * Tlen;
  float stored[8], S[8];
  const float st = startT[col];
#pragma unroll
  for (int r = 0; r < 8; ++r) {
    stored[r] = st + eb[(size_t)r * Tlen * Kdim];   // t = 0
    S[r] = 0.f;
  }

  // software pipeline: emissions/mask for step t prefetched during step t-1
  float emitNext[8];
  int   maskNext[8];
#pragma unroll
  for (int r = 0; r < 8; ++r) {
    emitNext[r] = eb[(size_t)r * Tlen * Kdim + Kdim];  // t = 1
    maskNext[r] = mb[(size_t)r * Tlen + 1];
  }

  for (int t = 1; t < Tlen; ++t) {
    float emitCur[8]; int maskCur[8];
#pragma unroll
    for (int r = 0; r < 8; ++r) { emitCur[r] = emitNext[r]; maskCur[r] = maskNext[r]; }
    if (t + 1 < Tlen) {
#pragma unroll
      for (int r = 0; r < 8; ++r) {
        emitNext[r] = eb[(size_t)r * Tlen * Kdim + (size_t)(t + 1) * Kdim];
        maskNext[r] = mb[(size_t)r * Tlen + (t + 1)];
      }
    }

    // per-row max over this wave's 16 columns (xor<16 stays inside half-wave)
    float lm[8];
#pragma unroll
    for (int r = 0; r < 8; ++r) lm[r] = stored[r];
#pragma unroll
    for (int d = 1; d < 16; d <<= 1) {
#pragma unroll
      for (int r = 0; r < 8; ++r) lm[r] = fmaxf(lm[r], __shfl_xor(lm[r], d, 32));
    }
    if (n == 0) {
#pragma unroll
      for (int r = 0; r < 8; ++r) ldsRed[r + off][wave] = lm[r];
    }
    __syncthreads();

    // combined per-row max across the 8 waves (128 columns)
    float M[8];
#pragma unroll
    for (int r = 0; r < 8; ++r) {
      const float4 u = *(const float4*)&ldsRed[r + off][0];
      const float4 v = *(const float4*)&ldsRed[r + off][4];
      M[r] = fmaxf(fmaxf(fmaxf(u.x, u.y), fmaxf(u.z, u.w)),
                   fmaxf(fmaxf(v.x, v.y), fmaxf(v.z, v.w)));
    }

    // stage A operand: V[row][col] = exp(stored - M) as f16  (arg <= 0, exact)
#pragma unroll
    for (int r = 0; r < 8; ++r)
      ldsA[r + off][col] = (_Float16)FAST_EXP2((stored[r] - M[r]) * LOG2E);
    __syncthreads();

    // gather all A fragments first (one dscnt wait), then chain the 4 WMMAs
    V16 a0, a1, a2, a3;
    a0.h[0] = *(const v8h*)&ldsA[n][0 + off];
    a0.h[1] = *(const v8h*)&ldsA[n][0 + off + 16];
    a1.h[0] = *(const v8h*)&ldsA[n][32 + off];
    a1.h[1] = *(const v8h*)&ldsA[n][32 + off + 16];
    a2.h[0] = *(const v8h*)&ldsA[n][64 + off];
    a2.h[1] = *(const v8h*)&ldsA[n][64 + off + 16];
    a3.h[0] = *(const v8h*)&ldsA[n][96 + off];
    a3.h[1] = *(const v8h*)&ldsA[n][96 + off + 16];

    v8f acc = {};
    acc = __builtin_amdgcn_wmma_f32_16x16x32_f16(false, a0.v, false, bm[0].v, (short)0, acc, false, false);
    acc = __builtin_amdgcn_wmma_f32_16x16x32_f16(false, a1.v, false, bm[1].v, (short)0, acc, false, false);
    acc = __builtin_amdgcn_wmma_f32_16x16x32_f16(false, a2.v, false, bm[2].v, (short)0, acc, false, false);
    acc = __builtin_amdgcn_wmma_f32_16x16x32_f16(false, a3.v, false, bm[3].v, (short)0, acc, false, false);

    // branchless masked alpha update: nxt = log(acc) + emit ; shift accumulates M
    // acc >= exp(0)*min(expT) ~ 1, so the raw v_log_f32 path is safe
#pragma unroll
    for (int r = 0; r < 8; ++r) {
      float nxt = FAST_LOG2(acc[r]) * LN2 + emitCur[r];
      stored[r] = maskCur[r] ? nxt : stored[r];
      S[r]      = maskCur[r] ? (S[r] + M[r]) : S[r];
    }
  }

  // ---- final logsumexp over j of (alpha + endT) ----
#pragma unroll
  for (int r = 0; r < 8; ++r) ldsF[r + off][col] = stored[r] + S[r];
  __syncthreads();
  if (wave == 0 && lane < 16) {
    const int row = lane;
    float m = -3.4e38f;
    for (int j = 0; j < Kdim; ++j) m = fmaxf(m, ldsF[row][j] + endT[j]);
    float s = 0.f;
    for (int j = 0; j < Kdim; ++j) s += FAST_EXP2((ldsF[row][j] + endT[j] - m) * LOG2E);
    out[b0 + row] = m + FAST_LOG2(s) * LN2 - gold[b0 + row];
  }
}

// ---------------------------------------------------------------------------
extern "C" void kernel_launch(void* const* d_in, const int* in_sizes, int n_in,
                              void* d_out, int out_size, void* d_ws, size_t ws_size,
                              hipStream_t stream) {
  const float* emissions   = (const float*)d_in[0];
  const float* transitions = (const float*)d_in[1];
  const float* startT      = (const float*)d_in[2];
  const float* endT        = (const float*)d_in[3];
  const int*   tags        = (const int*)d_in[4];
  const unsigned char* mask = (const unsigned char*)d_in[5];  // jnp bool -> u8
  float* out = (float*)d_out;

  _Float16* expT16 = (_Float16*)d_ws;                                   // 32 KB
  float* gold = (float*)((char*)d_ws + Kdim * Kdim * sizeof(_Float16)); // 1 KB

  crf_exp_transitions<<<(Kdim * Kdim + 255) / 256, 256, 0, stream>>>(transitions, expT16);
  crf_gold<<<Bsz / 8, 256, 0, stream>>>(emissions, transitions, startT, endT, tags, mask, gold);
  crf_logz<<<Bsz / 16, 256, 0, stream>>>(emissions, expT16, startT, endT, mask, gold, out);
}
